// MegaLayer_85220741087272
// MI455X (gfx1250) — matrix operand order repaired and checked by reading, not compiled, not run on hardware
//
#include <hip/hip_runtime.h>
#include <stdint.h>

#define BB 8
#define LL 2048
#define DD 128
#define HH 16
#define DQKN 64
#define DVN 256
#define MLT (BB*LL)   /* 16384 rows */

typedef __attribute__((ext_vector_type(16))) __bf16 v16bf;
typedef __attribute__((ext_vector_type(8)))  float  v8f;
typedef __attribute__((ext_vector_type(4)))  int    i32x4_;

union ABf { v16bf v; unsigned short u[16]; uint32_t w[8]; uint4 q[2]; };

#if defined(__gfx1250__) && __has_builtin(__builtin_amdgcn_global_load_async_to_lds_b128)
#define USE_ASYNC_LDS 1
#define GPTR4(p) ((__attribute__((address_space(1))) i32x4_*)(p))
#define LPTR4(p) ((__attribute__((address_space(3))) i32x4_*)(p))
#else
#define USE_ASYNC_LDS 0
#endif

__device__ __forceinline__ unsigned short f2bf(float f) {
  union { float f; uint32_t u; } c; c.f = f;
  uint32_t r = c.u + 0x7FFFu + ((c.u >> 16) & 1u);
  return (unsigned short)(r >> 16);
}
__device__ __forceinline__ float sigm_(float x) { return 1.0f / (1.0f + __expf(-x)); }
__device__ __forceinline__ float silu_(float x) { return x * sigm_(x); }
__device__ __forceinline__ v8f vzero8() { v8f v; for (int i = 0; i < 8; ++i) v[i] = 0.f; return v; }

// ---------------- EMA scan (exact closed form of the FFT conv) ----------------
__global__ void ema_kernel(const float* __restrict__ x,
                           const float* __restrict__ expansion,
                           const float* __restrict__ reduction,
                           const float* __restrict__ alphas,
                           const float* __restrict__ damps,
                           float* __restrict__ ema,
                           unsigned short* __restrict__ ema_bf,
                           unsigned short* __restrict__ x_bf) {
  const int b = blockIdx.x;
  const int d = threadIdx.x;     // 128 threads
  float c[HH], qf[HH], s[HH];
  for (int h = 0; h < HH; ++h) {
    float a  = sigm_(alphas[h]);
    float dp = sigm_(damps[h]);
    qf[h] = (1.f - a) * dp;
    c[h]  = a * expansion[h * DD + d] * reduction[h * DD + d];
    s[h]  = 0.f;
  }
  const float* xb = x + (size_t)b * LL * DD + d;
  float* eb = ema + (size_t)b * LL * DD + d;
  unsigned short* ebb = ema_bf + (size_t)b * LL * DD + d;
  unsigned short* xbb = x_bf  + (size_t)b * LL * DD + d;
  for (int n = 0; n < LL; ++n) {
    float xv = xb[(size_t)n * DD];
    float acc = 0.f;
#pragma unroll
    for (int h = 0; h < HH; ++h) { s[h] = qf[h] * s[h] + xv; acc += c[h] * s[h]; }
    eb[(size_t)n * DD]  = acc;
    ebb[(size_t)n * DD] = f2bf(acc);
    xbb[(size_t)n * DD] = f2bf(xv);
  }
}

// ---------------- T5 relative bias table (depends only on i-j >= 0) ----------
__global__ void bias_kernel(const float* __restrict__ rel_bias, float* __restrict__ table) {
  int n = blockIdx.x * blockDim.x + threadIdx.x;
  if (n >= LL) return;
  int bucket;
  if (n < 16) bucket = n;
  else {
    float v = __logf((float)n / 16.f) / __logf(128.f / 16.f) * 16.f;
    bucket = 16 + (int)v;
    if (bucket > 31) bucket = 31;
  }
  table[n] = rel_bias[bucket] * 8.0f;   // sqrt(DQK)=8
}

// -------- weight swizzle: fp32 KxN -> bf16 WMMA-B operand layout -------------
// dst[((kc*nnt + nt)*32 + lane)*16 + e] = bf16( W[(kc*32 + (lane>=16)*16 + e)*N + nt*16 + (lane&15)] )
__global__ void swz_kernel(const float* __restrict__ W, unsigned short* __restrict__ dst,
                           int K, int N) {
  int i = blockIdx.x * blockDim.x + threadIdx.x;
  if (i >= K * N) return;
  int e    = i & 15;
  int lane = (i >> 4) & 31;
  int rest = i >> 9;
  int nnt  = N >> 4;
  int nt   = rest % nnt;
  int kc   = rest / nnt;
  int kk   = kc * 32 + ((lane >> 4) << 4) + e;
  int col  = nt * 16 + (lane & 15);
  dst[i] = f2bf(W[(size_t)kk * N + col]);
}

// ---------------- generic bf16 WMMA GEMM: C = act(A@W + bias) ----------------
// block = 256 thr (8 waves); wave -> 16x64 tile; grid = (M/128, N/64)
// Wsw is in swizzled B-operand layout (see swz_kernel)
__global__ __launch_bounds__(256)
void gemm_kernel(const unsigned short* __restrict__ A,   // M x K bf16 row-major
                 const unsigned short* __restrict__ Wsw, // swizzled K x N bf16
                 const float* __restrict__ bias,         // N (or null)
                 int M, int N, int K, int mode,
                 const float* __restrict__ gamma,        // 2 x N (mode 1)
                 const float* __restrict__ beta,         // 2 x N (mode 1)
                 float* __restrict__ outf,
                 unsigned short* __restrict__ outb,
                 unsigned short* __restrict__ outb2) {
  const int lane = threadIdx.x & 31;
  const int wave = threadIdx.x >> 5;
  const int row0 = blockIdx.x * 128 + wave * 16;
  const int col0 = blockIdx.y * 64;
  const int m  = lane & 15;
  const int hi = (lane >> 4) & 1;
  const int nnt = N >> 4;
  v8f acc[4];
#pragma unroll
  for (int t = 0; t < 4; ++t) acc[t] = vzero8();
  const unsigned short* Arow = A + (size_t)(row0 + m) * K;
  for (int kc = 0; kc < K; kc += 32) {
    ABf a;
#pragma unroll
    for (int vg = 0; vg < 8; ++vg) {
      int k0 = kc + (hi << 4) + (vg << 1);
      a.w[vg] = *(const uint32_t*)(Arow + k0);
    }
#pragma unroll
    for (int t = 0; t < 4; ++t) {
      const unsigned short* wp =
          Wsw + ((((size_t)(kc >> 5) * nnt) + (col0 >> 4) + t) * 32 + lane) * 16;
      ABf bb;
      bb.q[0] = *(const uint4*)wp;
      bb.q[1] = *(const uint4*)(wp + 8);
      acc[t] = __builtin_amdgcn_wmma_f32_16x16x32_bf16(false, a.v, false, bb.v,
                                                       (short)0, acc[t], false, false);
    }
  }
#pragma unroll
  for (int t = 0; t < 4; ++t) {
    int col = col0 + t * 16 + m;
    float bv = bias ? bias[col] : 0.f;
#pragma unroll
    for (int r = 0; r < 8; ++r) {
      int row = row0 + r + hi * 8;
      float val = acc[t][r] + bv;
      size_t idx = (size_t)row * N + col;
      if (mode == 0)      { outb[idx] = f2bf(silu_(val)); }
      else if (mode == 1) {
        float s = silu_(val);
        outb[idx]  = f2bf(s * gamma[col]     + beta[col]);
        outb2[idx] = f2bf(s * gamma[N + col] + beta[N + col]);
      }
      else if (mode == 2) { outf[idx] = silu_(val); }
      else if (mode == 3) { outf[idx] = sigm_(val); }
      else                { outf[idx] = val; }
    }
  }
}

// ---------------- flash attention w/ causal mask + T5 bias + reset fuse ------
#define QTILE 128
#define KCH 32
__global__ __launch_bounds__(256)
void attn_kernel(const unsigned short* __restrict__ Q,   // B,L,DQK bf16
                 const unsigned short* __restrict__ Km,  // B,L,DQK bf16
                 const unsigned short* __restrict__ V,   // B,L,DV bf16
                 const float* __restrict__ btab,         // L
                 const float* __restrict__ reset,        // B,L,DV f32
                 unsigned short* __restrict__ gated) {   // B,L,DV bf16
  __shared__ __align__(16) unsigned short Klds[KCH * DQKN];   // [key][feat]
  __shared__ __align__(16) unsigned short Vlds[DVN * KCH];    // [dv][key] transposed
  __shared__ __align__(16) unsigned short Plds[8][16 * KCH];  // per-wave 16x32
  const int b   = blockIdx.y;
  const int qb0 = blockIdx.x * QTILE;
  const int tid = threadIdx.x;
  const int lane = tid & 31, wave = tid >> 5;
  const int m = lane & 15, hi = (lane >> 4) & 1;
  const int qrow0 = qb0 + wave * 16;
  const float scale = 0.08838834764831845f;   // 1/sqrt(D=128)

  // Q A-operand (register resident, 2 chunks of K=32)
  ABf qa[2];
  {
    const unsigned short* qrow = Q + ((size_t)b * LL + qrow0 + m) * DQKN;
#pragma unroll
    for (int c = 0; c < 2; ++c)
#pragma unroll
      for (int vg = 0; vg < 8; ++vg) {
        int k0 = c * 32 + (hi << 4) + (vg << 1);
        qa[c].w[vg] = *(const uint32_t*)(qrow + k0);
      }
  }
  v8f acc[16];
#pragma unroll
  for (int t = 0; t < 16; ++t) acc[t] = vzero8();
  float mrow[8], lrow[8];
#pragma unroll
  for (int r = 0; r < 8; ++r) { mrow[r] = -1e30f; lrow[r] = 0.f; }

  const int kend = qb0 + QTILE;
  for (int kb = 0; kb < kend; kb += KCH) {
    __syncthreads();
    { // K chunk: 32x64 bf16 = 4KB, one b128 per thread (async DMA when available)
      const unsigned short* ksrc = Km + ((size_t)b * LL + kb) * DQKN + tid * 8;
#if USE_ASYNC_LDS
      __builtin_amdgcn_global_load_async_to_lds_b128(GPTR4(ksrc), LPTR4(&Klds[tid * 8]),
                                                     0, 0);
#else
      *(uint4*)&Klds[tid * 8] = *(const uint4*)ksrc;
#endif
    }
    { // V chunk: 32x256 bf16 transposed into [dv][key]
      int key = tid >> 3;
      int dvb = (tid & 7) * 32;
      const unsigned short* vsrc = V + ((size_t)b * LL + kb + key) * DVN + dvb;
      __align__(16) unsigned short tmp[32];
#pragma unroll
      for (int i = 0; i < 4; ++i) *(uint4*)&tmp[i * 8] = *(const uint4*)(vsrc + i * 8);
#pragma unroll
      for (int i = 0; i < 32; ++i) Vlds[(dvb + i) * KCH + key] = tmp[i];
      // prefetch next chunk while we compute on this one
      if (kb + KCH < kend) {
        __builtin_prefetch(vsrc + (size_t)KCH * DVN, 0, 0);
        __builtin_prefetch(Km + ((size_t)b * LL + kb + KCH) * DQKN + tid * 8, 0, 0);
      }
    }
#if USE_ASYNC_LDS
#if __has_builtin(__builtin_amdgcn_s_wait_asynccnt)
    __builtin_amdgcn_s_wait_asynccnt(0);
#else
    asm volatile("s_wait_asynccnt 0" ::: "memory");
#endif
#endif
    __syncthreads();
    if (kb <= qrow0 + 15) {
      // S = Q K^T, two 16-key subtiles
      v8f s[2];
#pragma unroll
      for (int kt = 0; kt < 2; ++kt) {
        v8f sa = vzero8();
        int key = kt * 16 + m;
#pragma unroll
        for (int c = 0; c < 2; ++c) {
          ABf bb;
#pragma unroll
          for (int vg = 0; vg < 8; ++vg) {
            int f0 = c * 32 + (hi << 4) + (vg << 1);
            bb.w[vg] = *(const uint32_t*)&Klds[key * DQKN + f0];
          }
          sa = __builtin_amdgcn_wmma_f32_16x16x32_bf16(false, qa[c].v, false, bb.v,
                                                       (short)0, sa, false, false);
        }
        s[kt] = sa;
      }
      // scale + bias + strict causal mask
#pragma unroll
      for (int kt = 0; kt < 2; ++kt) {
        int j = kb + kt * 16 + m;
#pragma unroll
        for (int r = 0; r < 8; ++r) {
          int i = qrow0 + r + hi * 8;
          int delta = i - j;
          float v_ = s[kt][r] * scale;
          s[kt][r] = (delta >= 0) ? (v_ + btab[delta]) : -1e30f;
        }
      }
      // online softmax (16-lane half reductions match C-layout rows)
      float fac[8];
#pragma unroll
      for (int r = 0; r < 8; ++r) {
        float v_ = fmaxf(s[0][r], s[1][r]);
#pragma unroll
        for (int off = 1; off < 16; off <<= 1) v_ = fmaxf(v_, __shfl_xor(v_, off, 32));
        float mn = fmaxf(mrow[r], v_);
        float f  = (mrow[r] <= -1e29f) ? 0.f : __expf(mrow[r] - mn);
        float p0 = (s[0][r] <= -1e29f) ? 0.f : __expf(s[0][r] - mn);
        float p1 = (s[1][r] <= -1e29f) ? 0.f : __expf(s[1][r] - mn);
        s[0][r] = p0; s[1][r] = p1;
        float su = p0 + p1;
#pragma unroll
        for (int off = 1; off < 16; off <<= 1) su += __shfl_xor(su, off, 32);
        lrow[r] = lrow[r] * f + su;
        mrow[r] = mn;
        fac[r]  = f;
      }
#pragma unroll
      for (int t = 0; t < 16; ++t)
#pragma unroll
        for (int r = 0; r < 8; ++r) acc[t][r] *= fac[r];
      // P: C-layout -> LDS -> A-layout (16x32)
#pragma unroll
      for (int kt = 0; kt < 2; ++kt)
#pragma unroll
        for (int r = 0; r < 8; ++r)
          Plds[wave][(r + hi * 8) * KCH + kt * 16 + m] = f2bf(s[kt][r]);
      asm volatile("s_wait_dscnt 0" ::: "memory");
      ABf pa;
#pragma unroll
      for (int vg = 0; vg < 8; ++vg) {
        int k0 = (hi << 4) + (vg << 1);
        pa.w[vg] = *(const uint32_t*)&Plds[wave][m * KCH + k0];
      }
      // O += P @ V  (16 dv-tiles)
#pragma unroll
      for (int t = 0; t < 16; ++t) {
        ABf bb;
        int dv = t * 16 + m;
#pragma unroll
        for (int vg = 0; vg < 8; ++vg)
          bb.w[vg] = *(const uint32_t*)&Vlds[dv * KCH + (hi << 4) + (vg << 1)];
        acc[t] = __builtin_amdgcn_wmma_f32_16x16x32_bf16(false, pa.v, false, bb.v,
                                                         (short)0, acc[t], false, false);
      }
    }
  }
  // epilogue: normalize, fuse reset gate, emit bf16 for the Uh GEMM
#pragma unroll
  for (int r = 0; r < 8; ++r) {
    int row = qrow0 + r + hi * 8;
    float inv = 1.0f / lrow[r];
    const float* rrow = reset + ((size_t)b * LL + row) * DVN;
    unsigned short* grow = gated + ((size_t)b * LL + row) * DVN;
#pragma unroll
    for (int t = 0; t < 16; ++t) {
      int dv = t * 16 + m;
      grow[dv] = f2bf(acc[t][r] * inv * rrow[dv]);
    }
  }
}

// ---------------- final blend -------------------------------------------------
__global__ void final_kernel(const float* __restrict__ x,
                             const float* __restrict__ update,
                             const float* __restrict__ hpart,
                             const float* __restrict__ gu,
                             const float* __restrict__ bh,
                             float* __restrict__ out) {
  int i = blockIdx.x * blockDim.x + threadIdx.x;
  if (i >= MLT * DD) return;
  int d = i & (DD - 1);
  float h = silu_(hpart[i] + gu[i] + bh[d]);
  float u = update[i];
  out[i] = u * h + (1.f - u) * x[i];
}

extern "C" void kernel_launch(void* const* d_in, const int* in_sizes, int n_in,
                              void* d_out, int out_size, void* d_ws, size_t ws_size,
                              hipStream_t stream) {
  const float* x         = (const float*)d_in[0];
  const float* expansion = (const float*)d_in[1];
  const float* reduction = (const float*)d_in[2];
  const float* alphas    = (const float*)d_in[3];
  const float* dampens   = (const float*)d_in[4];
  const float* rel_bias  = (const float*)d_in[5];
  const float* W_qk      = (const float*)d_in[6];
  const float* b_qk      = (const float*)d_in[7];
  const float* gamma     = (const float*)d_in[8];
  const float* beta      = (const float*)d_in[9];
  const float* W_v       = (const float*)d_in[10];
  const float* b_v       = (const float*)d_in[11];
  const float* W_reset   = (const float*)d_in[12];
  const float* b_reset   = (const float*)d_in[13];
  const float* W_update  = (const float*)d_in[14];
  const float* b_update  = (const float*)d_in[15];
  const float* Wh        = (const float*)d_in[16];
  const float* Uh        = (const float*)d_in[17];
  const float* bh        = (const float*)d_in[18];
  float* out = (float*)d_out;

  char* wsp = (char*)d_ws;
  size_t off = 0;
  auto carve = [&](size_t bytes) -> void* {
    void* p = wsp + off;
    off = (off + bytes + 255) & ~(size_t)255;
    return p;
  };
  float*          ema_f   = (float*)carve((size_t)MLT * DD * 4);
  unsigned short* ema_bf  = (unsigned short*)carve((size_t)MLT * DD * 2);
  unsigned short* x_bf    = (unsigned short*)carve((size_t)MLT * DD * 2);
  unsigned short* q_bf    = (unsigned short*)carve((size_t)MLT * DQKN * 2);
  unsigned short* k_bf    = (unsigned short*)carve((size_t)MLT * DQKN * 2);
  unsigned short* v_bf    = (unsigned short*)carve((size_t)MLT * DVN * 2);
  float*          reset_f = (float*)carve((size_t)MLT * DVN * 4);
  float*          upd_f   = (float*)carve((size_t)MLT * DD * 4);
  float*          hpart_f = (float*)carve((size_t)MLT * DD * 4);
  unsigned short* gated_b = (unsigned short*)carve((size_t)MLT * DVN * 2);
  float*          gu_f    = (float*)carve((size_t)MLT * DD * 4);
  float*          btab    = (float*)carve((size_t)LL * 4);
  unsigned short* Wqk_b   = (unsigned short*)carve(DD * DQKN * 2);
  unsigned short* Wv_b    = (unsigned short*)carve(DD * DVN * 2);
  unsigned short* Wrst_b  = (unsigned short*)carve(DD * DVN * 2);
  unsigned short* Wupd_b  = (unsigned short*)carve(DD * DD * 2);
  unsigned short* Wh_b    = (unsigned short*)carve(DD * DD * 2);
  unsigned short* Uh_b    = (unsigned short*)carve(DVN * DD * 2);

  // 1. EMA scan + bf16 copies of ema and x
  ema_kernel<<<BB, DD, 0, stream>>>(x, expansion, reduction, alphas, dampens,
                                    ema_f, ema_bf, x_bf);
  // 2. T5 bias table
  bias_kernel<<<(LL + 255) / 256, 256, 0, stream>>>(rel_bias, btab);
  // 3. weights -> bf16 in WMMA B-operand layout
  swz_kernel<<<(DD * DQKN + 255) / 256, 256, 0, stream>>>(W_qk, Wqk_b, DD, DQKN);
  swz_kernel<<<(DD * DVN + 255) / 256, 256, 0, stream>>>(W_v, Wv_b, DD, DVN);
  swz_kernel<<<(DD * DVN + 255) / 256, 256, 0, stream>>>(W_reset, Wrst_b, DD, DVN);
  swz_kernel<<<(DD * DD + 255) / 256, 256, 0, stream>>>(W_update, Wupd_b, DD, DD);
  swz_kernel<<<(DD * DD + 255) / 256, 256, 0, stream>>>(Wh, Wh_b, DD, DD);
  swz_kernel<<<(DVN * DD + 255) / 256, 256, 0, stream>>>(Uh, Uh_b, DVN, DD);

  // 4. WMMA GEMMs
  gemm_kernel<<<dim3(MLT / 128, DQKN / 64), 256, 0, stream>>>(
      ema_bf, Wqk_b, b_qk, MLT, DQKN, DD, 1, gamma, beta, nullptr, q_bf, k_bf);
  gemm_kernel<<<dim3(MLT / 128, DVN / 64), 256, 0, stream>>>(
      x_bf, Wv_b, b_v, MLT, DVN, DD, 0, nullptr, nullptr, nullptr, v_bf, nullptr);
  gemm_kernel<<<dim3(MLT / 128, DVN / 64), 256, 0, stream>>>(
      ema_bf, Wrst_b, b_reset, MLT, DVN, DD, 2, nullptr, nullptr, reset_f, nullptr, nullptr);
  gemm_kernel<<<dim3(MLT / 128, DD / 64), 256, 0, stream>>>(
      ema_bf, Wupd_b, b_update, MLT, DD, DD, 3, nullptr, nullptr, upd_f, nullptr, nullptr);
  gemm_kernel<<<dim3(MLT / 128, DD / 64), 256, 0, stream>>>(
      ema_bf, Wh_b, nullptr, MLT, DD, DD, 4, nullptr, nullptr, hpart_f, nullptr, nullptr);

  // 5. flash attention (fused with reset gate -> gated bf16)
  attn_kernel<<<dim3(LL / QTILE, BB), 256, 0, stream>>>(q_bf, k_bf, v_bf, btab,
                                                        reset_f, gated_b);
  // 6. gated @ Uh
  gemm_kernel<<<dim3(MLT / 128, DD / 64), 256, 0, stream>>>(
      gated_b, Uh_b, nullptr, MLT, DD, DVN, 4, nullptr, nullptr, gu_f, nullptr, nullptr);
  // 7. final blend
  final_kernel<<<(MLT * DD + 255) / 256, 256, 0, stream>>>(x, upd_f, hpart_f, gu_f, bh, out);
}
